// Block_13615046328817
// MI455X (gfx1250) — matrix operand order repaired and checked by reading, not compiled, not run on hardware
//
#include <hip/hip_runtime.h>

// ---------------------------------------------------------------------------
// CDNA5 (gfx1250) transformer block: bf16 WMMA GEMMs + fused attention.
// wave32; v_wmma_f32_16x16x32_bf16 for every matmul; LDS-staged tiles with
// register double-buffering; attention uses zero-padded LDS tiles so all
// fragment loads/stores are branch-free (ds_load_b128 / ds_store_b128).
// ---------------------------------------------------------------------------

typedef __bf16 bf16_t;
typedef __attribute__((ext_vector_type(16))) __bf16 v16bf;
typedef __attribute__((ext_vector_type(8)))  float  v8f;

#define TOKENS 66560   // B*N = 1024*65
#define CDIM   1024
#define HEADS  16
#define HD     64
#define SEQ    65
#define SEQP   80      // SEQ padded to WMMA granularity
#define KP     96      // attn@v K padded to multiple of 32
#define LGS    84      // transposed-logits row stride (f32): 336B, 16B-mult
#define HIDDEN 4096

// ---------------- helpers ----------------

__device__ inline v8f wmma_bf16(v16bf a, v16bf b, v8f c) {
  return __builtin_amdgcn_wmma_f32_16x16x32_bf16(
      /*neg_a=*/false, a, /*neg_b=*/false, b,
      /*c_mod=*/(short)0, c, /*reuse_a=*/false, /*reuse_b=*/false);
}

// A-fragment (16x32, M x K), per ISA: lanes 0-15 row=lane K={0..7,16..23},
// lanes 16-31 row=lane-16 K={8..15,24..31}. Two 16B LDS loads per lane.
__device__ inline v16bf ld_frag_a(const bf16_t* base, int stride, int m, int lane) {
  const int kb = (lane >= 16) ? 8 : 0;
  const bf16_t* p = base + m * stride + kb;
  union { uint4 u[2]; v16bf v; } r;
  r.u[0] = *reinterpret_cast<const uint4*>(p);
  r.u[1] = *reinterpret_cast<const uint4*>(p + 16);
  return r.v;
}

// B-fragment (32x16, K x N): lane col = lane%16, lanes 0-15 K={0..7,16..23},
// lanes 16-31 K={8..15,24..31} (matches A-fragment; rows of W / k / vT).
__device__ inline v16bf ld_frag_b(const bf16_t* base, int stride, int n, int lane) {
  const int kb = (lane < 16) ? 0 : 8;
  const bf16_t* p = base + n * stride + kb;
  union { uint4 u[2]; v16bf v; } r;
  r.u[0] = *reinterpret_cast<const uint4*>(p);
  r.u[1] = *reinterpret_cast<const uint4*>(p + 16);
  return r.v;
}

// ---------------- f32 -> bf16 convert ----------------

__global__ __launch_bounds__(256) void cvt_bf16_kernel(
    const float* __restrict__ in, bf16_t* __restrict__ out, int n) {
  int i = blockIdx.x * 256 + threadIdx.x;
  if (i < n) out[i] = (bf16_t)in[i];
}

// ---------------- LayerNorm over C=1024, out bf16 ----------------

__global__ __launch_bounds__(256) void ln_kernel(
    const float* __restrict__ x, const float* __restrict__ g,
    const float* __restrict__ bta, bf16_t* __restrict__ out) {
  const int row = blockIdx.x;
  const int tid = threadIdx.x;
  const int lane = tid & 31;
  const int wid = tid >> 5;
  const float* xr = x + (size_t)row * CDIM;

  float v[4], s = 0.f, ss = 0.f;
#pragma unroll
  for (int j = 0; j < 4; ++j) {
    v[j] = xr[tid + j * 256];
    s += v[j];
    ss += v[j] * v[j];
  }
#pragma unroll
  for (int o = 16; o; o >>= 1) {
    s  += __shfl_xor(s, o);
    ss += __shfl_xor(ss, o);
  }
  __shared__ float ps[8], pss[8];
  if (lane == 0) { ps[wid] = s; pss[wid] = ss; }
  __syncthreads();
  s = 0.f; ss = 0.f;
#pragma unroll
  for (int w = 0; w < 8; ++w) { s += ps[w]; ss += pss[w]; }
  const float mu  = s * (1.0f / CDIM);
  const float var = ss * (1.0f / CDIM) - mu * mu;
  const float inv = rsqrtf(var + 1e-5f);
#pragma unroll
  for (int j = 0; j < 4; ++j) {
    int c = tid + j * 256;
    out[(size_t)row * CDIM + c] = (bf16_t)((v[j] - mu) * inv * g[c] + bta[c]);
  }
}

// ---------------- tiled bf16 WMMA GEMM: out = epi(A @ W^T + bias) ----------
// A: [M,K] bf16 row-major, W: [Nn,K] bf16 row-major. 256 thr = 8 waves,
// block tile 128x128x32; wave tile 64x32 (4x2 wmma frags); register
// double-buffered global->LDS staging.
// MODE 0: qkv scatter->q/k/v[B,H,N,D] bf16   MODE 1: +resid -> f32
// MODE 2: GELU(tanh) -> bf16                 MODE 3: +resid -> f32

#define BM 128
#define BN 128
#define BK 32
#define LDT 40   // padded LDS row stride (bf16): 80B, 16B-multiple

template <int MODE>
__global__ __launch_bounds__(256) void gemm_kernel(
    const bf16_t* __restrict__ A, const bf16_t* __restrict__ W,
    const float* __restrict__ bias, const float* __restrict__ resid,
    float* __restrict__ outf, bf16_t* __restrict__ outb,
    bf16_t* __restrict__ qout, bf16_t* __restrict__ kout,
    bf16_t* __restrict__ vout, int K, int Nn) {
  __shared__ __align__(16) bf16_t As[BM * LDT];
  __shared__ __align__(16) bf16_t Bs[BN * LDT];

  const int tid  = threadIdx.x;
  const int lane = tid & 31;
  const int wid  = tid >> 5;
  const int wm   = wid & 1;    // 2 waves along M (64 rows each)
  const int wn   = wid >> 1;   // 4 waves along N (32 cols each)
  const int blockM = blockIdx.y * BM;
  const int blockN = blockIdx.x * BN;

  v8f acc[4][2];
#pragma unroll
  for (int mi = 0; mi < 4; ++mi)
#pragma unroll
    for (int ni = 0; ni < 2; ++ni) {
      v8f z = {0.f, 0.f, 0.f, 0.f, 0.f, 0.f, 0.f, 0.f};
      acc[mi][ni] = z;
    }

  const int ldr = tid >> 2;        // 0..63
  const int ldk = (tid & 3) * 8;   // 0,8,16,24
  const bf16_t* Ap0 = A + (size_t)(blockM + ldr) * K + ldk;
  const bf16_t* Ap1 = A + (size_t)(blockM + ldr + 64) * K + ldk;
  const bf16_t* Wp0 = W + (size_t)(blockN + ldr) * K + ldk;
  const bf16_t* Wp1 = W + (size_t)(blockN + ldr + 64) * K + ldk;

  // prime the register staging buffer
  uint4 ra0 = *reinterpret_cast<const uint4*>(Ap0);
  uint4 ra1 = *reinterpret_cast<const uint4*>(Ap1);
  uint4 rb0 = *reinterpret_cast<const uint4*>(Wp0);
  uint4 rb1 = *reinterpret_cast<const uint4*>(Wp1);

  for (int kk = 0; kk < K; kk += BK) {
    *reinterpret_cast<uint4*>(&As[ldr * LDT + ldk])        = ra0;
    *reinterpret_cast<uint4*>(&As[(ldr + 64) * LDT + ldk]) = ra1;
    *reinterpret_cast<uint4*>(&Bs[ldr * LDT + ldk])        = rb0;
    *reinterpret_cast<uint4*>(&Bs[(ldr + 64) * LDT + ldk]) = rb1;
    __syncthreads();

    // prefetch next k-step while WMMAs run on the current tiles
    if (kk + BK < K) {
      ra0 = *reinterpret_cast<const uint4*>(Ap0 + kk + BK);
      ra1 = *reinterpret_cast<const uint4*>(Ap1 + kk + BK);
      rb0 = *reinterpret_cast<const uint4*>(Wp0 + kk + BK);
      rb1 = *reinterpret_cast<const uint4*>(Wp1 + kk + BK);
    }

    v16bf af[4], bfr[2];
#pragma unroll
    for (int mi = 0; mi < 4; ++mi)
      af[mi] = ld_frag_a(As, LDT, wm * 64 + mi * 16 + (lane & 15), lane);
#pragma unroll
    for (int ni = 0; ni < 2; ++ni)
      bfr[ni] = ld_frag_b(Bs, LDT, wn * 32 + ni * 16 + (lane & 15), lane);
#pragma unroll
    for (int mi = 0; mi < 4; ++mi)
#pragma unroll
      for (int ni = 0; ni < 2; ++ni)
        acc[mi][ni] = wmma_bf16(af[mi], bfr[ni], acc[mi][ni]);
    __syncthreads();
  }

  // epilogue; C/D layout: VGPR i -> row = i (+8 for lanes 16-31), col = lane%16
  const int hi8 = (lane >> 4) << 3;
  const int cn  = lane & 15;
#pragma unroll
  for (int mi = 0; mi < 4; ++mi) {
    const int grow0 = blockM + wm * 64 + mi * 16 + hi8;
#pragma unroll
    for (int ni = 0; ni < 2; ++ni) {
      const int gcol = blockN + wn * 32 + ni * 16 + cn;
      const float bcol = bias[gcol];
      if (MODE == 0) {
        // scatter qkv col (3,H,D), row (B,N) -> [B,H,N,D]; incremental /65
        const int part = gcol >> 10;
        const int rem  = gcol & 1023;
        const int hh   = rem >> 6;
        const int dd   = rem & 63;
        int bb = grow0 / SEQ;
        int nn = grow0 - bb * SEQ;
        bf16_t* dst = (part == 0) ? qout : (part == 1) ? kout : vout;
#pragma unroll
        for (int i = 0; i < 8; ++i) {
          const size_t idx = (((size_t)(bb * HEADS + hh)) * SEQ + nn) * HD + dd;
          dst[idx] = (bf16_t)(acc[mi][ni][i] + bcol);
          if (++nn == SEQ) { nn = 0; ++bb; }
        }
      } else if (MODE == 2) {
        size_t idx = (size_t)grow0 * Nn + gcol;
#pragma unroll
        for (int i = 0; i < 8; ++i) {
          const float val = acc[mi][ni][i] + bcol;
          const float u = 0.7978845608028654f * (val + 0.044715f * val * val * val);
          outb[idx] = (bf16_t)(0.5f * val * (1.0f + tanhf(u)));
          idx += Nn;
        }
      } else {  // MODE 1 / 3: residual add, f32 out
        size_t idx = (size_t)grow0 * Nn + gcol;
#pragma unroll
        for (int i = 0; i < 8; ++i) {
          outf[idx] = resid[idx] + acc[mi][ni][i] + bcol;
          idx += Nn;
        }
      }
    }
  }
}

// ---------------- fused attention per (b,h): softmax(q k^T/8 + bias) v -----
// All LDS tiles zero-padded to WMMA granularity -> branch-free ds_load_b128
// fragment loads. Logits stored TRANSPOSED (col-major) so each C-fragment
// writes 2x ds_store_b128, no guards. Bias+scale applied in softmax phase.

__global__ __launch_bounds__(256) void attn_kernel(
    const bf16_t* __restrict__ qg, const bf16_t* __restrict__ kg,
    const bf16_t* __restrict__ vg, const float* __restrict__ abias,
    const float* __restrict__ bscale_p, bf16_t* __restrict__ og) {
  __shared__ __align__(16) bf16_t qs[SEQP * HD];      // 80x64, rows 65..79 = 0
  __shared__ __align__(16) bf16_t ks2[SEQP * HD];     // 80x64, rows 65..79 = 0
  __shared__ __align__(16) bf16_t vt[HD * KP];        // v^T 64x96, cols 65..95 = 0
  __shared__ __align__(16) float  lgT[SEQP * LGS];    // logits^T [col][row] 80x84
  __shared__ __align__(16) bf16_t at[SEQP * KP];      // attn 80x96, pads = 0

  const int bh  = blockIdx.x;
  const int bb  = bh >> 4;
  const int hh  = bh & 15;
  const int tid = threadIdx.x;
  const int lane = tid & 31;
  const int wid  = tid >> 5;

  // stage q,k (4160 bf16 each = 520 uint4) and zero pad rows (960 bf16 = 120 uint4)
  const uint4* gq = reinterpret_cast<const uint4*>(qg + (size_t)bh * SEQ * HD);
  const uint4* gk = reinterpret_cast<const uint4*>(kg + (size_t)bh * SEQ * HD);
  uint4* sq = reinterpret_cast<uint4*>(qs);
  uint4* sk = reinterpret_cast<uint4*>(ks2);
  for (int i = tid; i < 520; i += 256) { sq[i] = gq[i]; sk[i] = gk[i]; }
  const uint4 z4 = {0u, 0u, 0u, 0u};
  for (int i = 520 + tid; i < 640; i += 256) { sq[i] = z4; sk[i] = z4; }
  // stage v transposed (B-fragments of attn@v become K-contiguous)
  const bf16_t* gv = vg + (size_t)bh * SEQ * HD;
  for (int i = tid; i < SEQ * HD; i += 256) {
    const int n2 = i >> 6, dd = i & 63;
    vt[dd * KP + n2] = gv[i];
  }
  for (int i = tid; i < HD * 31; i += 256) {   // zero K-pad cols 65..95
    const int dd = i / 31, n2 = SEQ + (i % 31);
    vt[dd * KP + n2] = (bf16_t)0.0f;
  }
  __syncthreads();

  const int hi8 = (lane >> 4) << 3;
  const int cn  = lane & 15;

  // phase 1: raw logits = q k^T (5x5 tiles over 8 waves); the 8 accumulator
  // elements are 8 consecutive rows at one column -> transposed store is two
  // contiguous 16B ds_store_b128, unguarded (pad rows of lgT never read).
  for (int t = wid; t < 25; t += 8) {
    const int mt = t / 5, nt = t % 5;
    v8f acc = {0.f, 0.f, 0.f, 0.f, 0.f, 0.f, 0.f, 0.f};
#pragma unroll
    for (int kk = 0; kk < HD; kk += 32) {
      v16bf a  = ld_frag_a(qs + kk, HD, mt * 16 + cn, lane);
      v16bf bB = ld_frag_b(ks2 + kk, HD, nt * 16 + cn, lane);
      acc = wmma_bf16(a, bB, acc);
    }
    const int c  = nt * 16 + cn;
    const int r0 = mt * 16 + hi8;
    union { v8f v; uint4 u[2]; } cvt;
    cvt.v = acc;
    uint4* dst = reinterpret_cast<uint4*>(&lgT[c * LGS + r0]);
    dst[0] = cvt.u[0];
    dst[1] = cvt.u[1];
  }
  __syncthreads();

  // phase 2: scale + bias + row softmax (wave32 shuffles, coalesced bias reads)
  const float bscale = bscale_p[0];
  const float* brow = abias + (size_t)hh * (SEQ * SEQ);
  for (int r = wid; r < SEQ; r += 8) {
    const float v0 = lgT[lane * LGS + r] * 0.125f + bscale * brow[r * SEQ + lane];
    const float v1 =
        lgT[(lane + 32) * LGS + r] * 0.125f + bscale * brow[r * SEQ + lane + 32];
    const float v2 = (lane == 0)
        ? lgT[64 * LGS + r] * 0.125f + bscale * brow[r * SEQ + 64]
        : -3.0e38f;
    float m = fmaxf(v0, fmaxf(v1, v2));
#pragma unroll
    for (int o = 16; o; o >>= 1) m = fmaxf(m, __shfl_xor(m, o));
    const float e0 = __expf(v0 - m);
    const float e1 = __expf(v1 - m);
    const float e2 = (lane == 0) ? __expf(v2 - m) : 0.f;
    float s = e0 + e1 + e2;
#pragma unroll
    for (int o = 16; o; o >>= 1) s += __shfl_xor(s, o);
    const float inv = 1.0f / s;
    at[r * KP + lane]      = (bf16_t)(e0 * inv);
    at[r * KP + lane + 32] = (bf16_t)(e1 * inv);
    at[r * KP + lane + 64] = (bf16_t)(e2 * inv);   // cols 64..95 (pads -> 0)
  }
  // zero pad rows 65..79 of attn so phase-3 A fragments are branch-free
  for (int i = tid; i < (SEQP - SEQ) * KP; i += 256) at[SEQ * KP + i] = (bf16_t)0.0f;
  __syncthreads();

  // phase 3: o = attn @ v; tiles mt<4 (rows <=63) store unguarded; the mt==4
  // tile has only row 64 real -> single masked store of acc[0].
  const size_t obase = (size_t)bb * SEQ * CDIM + (size_t)hh * HD;
  for (int t = wid; t < 16; t += 8) {
    const int mt = t >> 2, nt = t & 3;
    v8f acc = {0.f, 0.f, 0.f, 0.f, 0.f, 0.f, 0.f, 0.f};
#pragma unroll
    for (int kk = 0; kk < KP; kk += 32) {
      v16bf a  = ld_frag_a(at + kk, KP, mt * 16 + cn, lane);
      v16bf bB = ld_frag_b(vt + kk, KP, nt * 16 + cn, lane);
      acc = wmma_bf16(a, bB, acc);
    }
    const int dd = nt * 16 + cn;
    size_t oidx = obase + (size_t)(mt * 16 + hi8) * CDIM + dd;
#pragma unroll
    for (int i = 0; i < 8; ++i) {
      og[oidx] = (bf16_t)acc[i];
      oidx += CDIM;
    }
  }
  if (wid < 4) {  // mt == 4 boundary tile
    const int nt = wid;
    v8f acc = {0.f, 0.f, 0.f, 0.f, 0.f, 0.f, 0.f, 0.f};
#pragma unroll
    for (int kk = 0; kk < KP; kk += 32) {
      v16bf a  = ld_frag_a(at + kk, KP, 64 + cn, lane);
      v16bf bB = ld_frag_b(vt + kk, KP, nt * 16 + cn, lane);
      acc = wmma_bf16(a, bB, acc);
    }
    if (hi8 == 0)  // acc[0] of lanes 0-15 holds row 64
      og[obase + (size_t)64 * CDIM + nt * 16 + cn] = (bf16_t)acc[0];
  }
}

// ---------------- launch ----------------

extern "C" void kernel_launch(void* const* d_in, const int* in_sizes, int n_in,
                              void* d_out, int out_size, void* d_ws, size_t ws_size,
                              hipStream_t stream) {
  (void)in_sizes; (void)n_in; (void)out_size; (void)ws_size;
  const float* x      = (const float*)d_in[0];
  const float* ln1_g  = (const float*)d_in[1];
  const float* ln1_b  = (const float*)d_in[2];
  const float* qkv_w  = (const float*)d_in[3];
  const float* qkv_b  = (const float*)d_in[4];
  const float* proj_w = (const float*)d_in[5];
  const float* proj_b = (const float*)d_in[6];
  const float* abias  = (const float*)d_in[7];
  const float* bscale = (const float*)d_in[8];
  const float* ln2_g  = (const float*)d_in[9];
  const float* ln2_b  = (const float*)d_in[10];
  const float* fc1_w  = (const float*)d_in[11];
  const float* fc1_b  = (const float*)d_in[12];
  const float* fc2_w  = (const float*)d_in[13];
  const float* fc2_b  = (const float*)d_in[14];
  float* out = (float*)d_out;

  char* ws = (char*)d_ws;
  size_t off = 0;
  auto alloc = [&](size_t bytes) -> char* {
    char* p = ws + off;
    off += (bytes + 255) & ~(size_t)255;
    return p;
  };
  const size_t T = TOKENS;
  bf16_t* hbuf = (bf16_t*)alloc(T * CDIM * 2);             // LN1 out / LN2 out
  bf16_t* qw   = (bf16_t*)alloc((size_t)3 * CDIM * CDIM * 2);
  bf16_t* pw   = (bf16_t*)alloc((size_t)CDIM * CDIM * 2);
  bf16_t* f1w  = (bf16_t*)alloc((size_t)HIDDEN * CDIM * 2);
  bf16_t* f2w  = (bf16_t*)alloc((size_t)CDIM * HIDDEN * 2);
  bf16_t* qb   = (bf16_t*)alloc(T * CDIM * 2);             // [B,H,N,D]
  bf16_t* kb   = (bf16_t*)alloc(T * CDIM * 2);
  bf16_t* vb   = (bf16_t*)alloc(T * CDIM * 2);
  bf16_t* ob   = (bf16_t*)alloc(T * CDIM * 2);             // attn out [B,N,C]
  float*  x2   = (float*)alloc(T * CDIM * 4);              // post-attn residual
  bf16_t* abuf = qb;  // MLP hidden (T*HIDDEN bf16) aliases dead q/k/v/o region

  // weight converts (f32 -> bf16; L2-resident afterwards)
  cvt_bf16_kernel<<<(3 * CDIM * CDIM + 255) / 256, 256, 0, stream>>>(qkv_w, qw, 3 * CDIM * CDIM);
  cvt_bf16_kernel<<<(CDIM * CDIM + 255) / 256, 256, 0, stream>>>(proj_w, pw, CDIM * CDIM);
  cvt_bf16_kernel<<<(HIDDEN * CDIM + 255) / 256, 256, 0, stream>>>(fc1_w, f1w, HIDDEN * CDIM);
  cvt_bf16_kernel<<<(CDIM * HIDDEN + 255) / 256, 256, 0, stream>>>(fc2_w, f2w, CDIM * HIDDEN);

  // LN1
  ln_kernel<<<TOKENS, 256, 0, stream>>>(x, ln1_g, ln1_b, hbuf);
  // QKV GEMM + scatter to [B,H,N,D]
  gemm_kernel<0><<<dim3(3 * CDIM / BN, TOKENS / BM), 256, 0, stream>>>(
      hbuf, qw, qkv_b, nullptr, nullptr, nullptr, qb, kb, vb, CDIM, 3 * CDIM);
  // fused attention
  attn_kernel<<<1024 * HEADS, 256, 0, stream>>>(qb, kb, vb, abias, bscale, ob);
  // proj + residual -> x2 (f32)
  gemm_kernel<1><<<dim3(CDIM / BN, TOKENS / BM), 256, 0, stream>>>(
      ob, pw, proj_b, x, x2, nullptr, nullptr, nullptr, nullptr, CDIM, CDIM);
  // LN2 (reuse hbuf)
  ln_kernel<<<TOKENS, 256, 0, stream>>>(x2, ln2_g, ln2_b, hbuf);
  // FC1 + GELU -> abuf (bf16)
  gemm_kernel<2><<<dim3(HIDDEN / BN, TOKENS / BM), 256, 0, stream>>>(
      hbuf, f1w, fc1_b, nullptr, nullptr, abuf, nullptr, nullptr, nullptr, CDIM, HIDDEN);
  // FC2 + residual -> out (f32)
  gemm_kernel<3><<<dim3(CDIM / BN, TOKENS / BM), 256, 0, stream>>>(
      abuf, f2w, fc2_b, x2, out, nullptr, nullptr, nullptr, nullptr, HIDDEN, CDIM);
}
